// DentalAnatomyLoss_17411797418431
// MI455X (gfx1250) — compile-verified, matrix operand order checked by reading, not run on hardware
//
#include <hip/hip_runtime.h>
#include <hip/hip_bf16.h>

typedef __attribute__((ext_vector_type(16))) _Float16 v16h;
typedef __attribute__((ext_vector_type(8)))  float    v8f;
typedef __attribute__((ext_vector_type(16))) float    v16f;

#define DW   128                 // W
#define DH   128                 // H
#define DD   64                  // D
#define SLICE (DH * DW)          // 16384 elements per z-slice
#define NCH   64                 // B*C channels
#define NSLICE (NCH * DD)        // 4096 blocks

// ws layout (floats): [0..63] crown per channel, [64..127] root per channel,
// [128] sum|dx|, [129] sum|dy|, [130] sum|dz|
#define WS_GX 128
#define WS_GY 129
#define WS_GZ 130
#define WS_N  131

__device__ __forceinline__ v8f wmma_rowsum_acc(v16h a, v16h ones, v8f c) {
  // D = A(16x32 f16) x ones(32x16) + C  ->  D[m][n] = rowsum(A,m), f32 accumulate
  return __builtin_amdgcn_wmma_f32_16x16x32_f16(
      /*neg_a=*/false, a, /*neg_b=*/false, ones,
      /*c_mod=*/(short)0, c, /*reuse_a=*/false, /*reuse_b=*/false);
}

__global__ void anatomy_zero_ws(float* __restrict__ wsf) {
  int t = threadIdx.x;
  if (t < WS_N) wsf[t] = 0.0f;
}

__global__ __launch_bounds__(256) void anatomy_reduce(
    const float* __restrict__ seg, float* __restrict__ wsf) {
  const int slice = blockIdx.x;        // 0..4095
  const int ch = slice >> 6;           // channel (b*C + c)
  const int d  = slice & 63;           // z index within channel
  const size_t base0 = (size_t)slice * SLICE;
  const bool hasZ = (d < DD - 1);

  v16h ones;
#pragma unroll
  for (int j = 0; j < 16; ++j) ones[j] = (_Float16)1.0f;

  v8f accV = {}, accX = {}, accY = {}, accZ = {};

#pragma unroll
  for (int i = 0; i < 4; ++i) {
    const int s  = i * 256 + threadIdx.x;   // segment 0..1023 within slice
    const int h  = s >> 3;                  // row 0..127
    const int w0 = (s & 7) << 4;            // 0,16,...,112
    const size_t base = base0 + (size_t)s * 16;

    if (i < 3) __builtin_prefetch((const void*)(seg + base + 4096), 0, 1);

    const v16f av = *(const v16f*)(seg + base);

    // ---- value sum (crown/root) ----
    v16h va;
#pragma unroll
    for (int j = 0; j < 16; ++j) va[j] = (_Float16)av[j];
    accV = wmma_rowsum_acc(va, ones, accV);

    // ---- |dx| : neighbor at +1 (within row), gated at w==127 ----
    const bool lastInRow = (w0 == DW - 16);
    float nxt = lastInRow ? 0.0f : seg[base + 16];
    v16h vx;
#pragma unroll
    for (int j = 0; j < 15; ++j) vx[j] = (_Float16)fabsf(av[j + 1] - av[j]);
    vx[15] = lastInRow ? (_Float16)0.0f : (_Float16)fabsf(nxt - av[15]);
    accX = wmma_rowsum_acc(vx, ones, accX);

    // ---- |dy| : neighbor at +W, gated at h==127 ----
    v16h vy;
    if (h < DH - 1) {
      const v16f by = *(const v16f*)(seg + base + DW);
#pragma unroll
      for (int j = 0; j < 16; ++j) vy[j] = (_Float16)fabsf(by[j] - av[j]);
    } else {
#pragma unroll
      for (int j = 0; j < 16; ++j) vy[j] = (_Float16)0.0f;
    }
    accY = wmma_rowsum_acc(vy, ones, accY);

    // ---- |dz| : neighbor at +H*W, gated at d==63 (block-uniform) ----
    v16h vz;
    if (hasZ) {
      const v16f bz = *(const v16f*)(seg + base + SLICE);
#pragma unroll
      for (int j = 0; j < 16; ++j) vz[j] = (_Float16)fabsf(bz[j] - av[j]);
    } else {
#pragma unroll
      for (int j = 0; j < 16; ++j) vz[j] = (_Float16)0.0f;
    }
    accZ = wmma_rowsum_acc(vz, ones, accZ);
  }

  // Per-lane sum of the 8 C-matrix VGPRs. With B = ones, D[m][n] is identical
  // for all n, so lanes 0-15 all hold sum(rows 0..7) and lanes 16-31 all hold
  // sum(rows 8..15); lanes 0 and 16 together cover the full wave total.
  float pV = 0.f, pX = 0.f, pY = 0.f, pZ = 0.f;
#pragma unroll
  for (int r = 0; r < 8; ++r) {
    pV += accV[r]; pX += accX[r]; pY += accY[r]; pZ += accZ[r];
  }

  __shared__ float red[4];
  if (threadIdx.x < 4) red[threadIdx.x] = 0.0f;
  __syncthreads();
  const int lane = threadIdx.x & 31;
  if (lane == 0 || lane == 16) {
    atomicAdd(&red[0], pV);
    atomicAdd(&red[1], pX);
    atomicAdd(&red[2], pY);
    atomicAdd(&red[3], pZ);
  }
  __syncthreads();
  if (threadIdx.x == 0) {
    const int bin = (d < DD / 2) ? ch : (NCH + ch);
    atomicAdd(&wsf[bin],   red[0]);
    atomicAdd(&wsf[WS_GX], red[1]);
    atomicAdd(&wsf[WS_GY], red[2]);
    atomicAdd(&wsf[WS_GZ], red[3]);
  }
}

__global__ void anatomy_finalize(const float* __restrict__ wsf,
                                 float* __restrict__ out) {
  __shared__ float terms[NCH];
  const int t = threadIdx.x;
  if (t < NCH) {
    float c = wsf[t], r = wsf[NCH + t];
    float tot = c + r;
    float term = 0.0f;
    if (tot > 0.0f && r > 0.0f) {
      float ratio = c / r - 1.2f;   // EXPECTED_RATIO
      term = ratio * ratio;
    }
    terms[t] = term;
  }
  __syncthreads();
  if (t == 0) {
    float s = 0.0f;
#pragma unroll
    for (int j = 0; j < NCH; ++j) s += terms[j];
    const float cr_loss = s / (float)NCH;
    // grad means: counts are B*C*D*H*(W-1), B*C*D*(H-1)*W, B*C*(D-1)*H*W
    const float mx = wsf[WS_GX] / 66584576.0f;  // 2*32*64*128*127
    const float my = wsf[WS_GY] / 66584576.0f;  // 2*32*64*127*128
    const float mz = wsf[WS_GZ] / 66060288.0f;  // 2*32*63*128*128
    const float crown_root = 2.0f * cr_loss;          // CROWN_ROOT_W
    const float smooth     = 1.5f * (mx + my + mz);   // SMOOTH_W
    out[0] = crown_root;
    out[1] = smooth;
    out[2] = crown_root + smooth;
  }
}

extern "C" void kernel_launch(void* const* d_in, const int* in_sizes, int n_in,
                              void* d_out, int out_size, void* d_ws, size_t ws_size,
                              hipStream_t stream) {
  const float* seg = (const float*)d_in[0];
  float* wsf = (float*)d_ws;
  float* out = (float*)d_out;
  (void)in_sizes; (void)n_in; (void)out_size; (void)ws_size;

  anatomy_zero_ws<<<1, 192, 0, stream>>>(wsf);
  anatomy_reduce<<<NSLICE, 256, 0, stream>>>(seg, wsf);
  anatomy_finalize<<<1, NCH, 0, stream>>>(wsf, out);
}
